// nBRC_13176959664608
// MI455X (gfx1250) — matrix operand order repaired and checked by reading, not compiled, not run on hardware
//
#include <hip/hip_runtime.h>

// Problem dims (fixed by the reference)
#define TT 512
#define BB 64
#define II 512
#define HH 1024
#define TBM (TT * BB)     // 32768 rows in the projection GEMMs
#define LDST 40           // LDS row stride in bf16 elems (32 + 8 pad, 80B rows)

// ---- CDNA5 async-to-LDS path (guarded; falls back to sync b128 copies) ----
#if defined(__gfx1250__) && \
    __has_builtin(__builtin_amdgcn_global_load_async_to_lds_b128) && \
    __has_builtin(__builtin_amdgcn_s_wait_asynccnt)
#define NBRC_ASYNC 1
#else
#define NBRC_ASYNC 0
#endif

typedef __attribute__((ext_vector_type(16))) __bf16 bf16x16;
typedef __attribute__((ext_vector_type(8)))  float  f32x8;
typedef __attribute__((ext_vector_type(4)))  int    v4i;

#if NBRC_ASYNC
typedef __attribute__((address_space(1))) v4i* gv4i_ptr;  // global (AS1)
typedef __attribute__((address_space(3))) v4i* lv4i_ptr;  // LDS (AS3)
#endif

union AB16 { bf16x16 v; unsigned int u[8]; };
union C32  { f32x8 v; float f[8]; };

__device__ __forceinline__ unsigned short f2bf(float x) {
  unsigned int u = __float_as_uint(x);
  u += 0x7FFFu + ((u >> 16) & 1u);      // round-to-nearest-even
  return (unsigned short)(u >> 16);
}
__device__ __forceinline__ unsigned int pack2bf(float a, float b) {
  return (unsigned int)f2bf(a) | ((unsigned int)f2bf(b) << 16);
}

// 16B global -> LDS copy: async (ASYNCcnt) on CDNA5, sync fallback otherwise.
__device__ __forceinline__ void copy_b128_to_lds(const void* g, void* l) {
#if NBRC_ASYNC
  __builtin_amdgcn_global_load_async_to_lds_b128(
      (gv4i_ptr)const_cast<void*>(g), (lv4i_ptr)l, 0, 0);
#else
  uint4 v = *reinterpret_cast<const uint4*>(g);
  *reinterpret_cast<uint4*>(l) = v;
#endif
}

template <int N>
__device__ __forceinline__ void wait_async() {
#if NBRC_ASYNC
  __builtin_amdgcn_s_wait_asynccnt(N);
#endif
}

// ---------------------------------------------------------------------------
// Kernel 0: elementwise fp32 -> bf16 (packs pairs; all our sizes are even)
// ---------------------------------------------------------------------------
__global__ __launch_bounds__(256)
void f32_to_bf16_kernel(const float* __restrict__ in,
                        unsigned int* __restrict__ out, int npairs)
{
  int i = blockIdx.x * 256 + threadIdx.x;
  if (i < npairs) {
    float2 v = *reinterpret_cast<const float2*>(in + 2 * (size_t)i);
    out[i] = pack2bf(v.x, v.y);
  }
}

// ---------------------------------------------------------------------------
// Kernel 1: projection GEMM  out[TBM,HH] = Xb[TBM,II](bf16) @ Ub[HH,II]^T + b
// Block: 256 thr (8 waves), tile 128M x 64N, K in 32-chunks, double-buffered
// LDS staged via async-to-LDS. Each wave: 32x32 macrotile = 2x2 WMMA tiles.
// ---------------------------------------------------------------------------
__global__ __launch_bounds__(256)
void proj_gemm_kernel(const unsigned short* __restrict__ Xb,
                      const unsigned short* __restrict__ Ub,
                      const float* __restrict__ bias,
                      float* __restrict__ out)
{
  __shared__ unsigned short As[2][128 * LDST];   // 128x32 bf16, padded
  __shared__ unsigned short Bs[2][64 * LDST];    // 64x32 bf16, padded

  const int tid   = threadIdx.x;
  const int lane  = tid & 31;
  const int wid   = tid >> 5;
  const int waveM = wid >> 1;          // 0..3
  const int waveN = wid & 1;           // 0..1
  const int blockM = blockIdx.x * 128;
  const int blockN = blockIdx.y * 64;
  const int laneHalf = lane >> 4;      // 0 or 1
  const int laneRow  = lane & 15;
  const int srow = tid >> 2;           // 0..63 staging row
  const int sseg = tid & 3;            // 0..3  (8 halfs / 16B per segment)

  C32 acc[2][2];
  #pragma unroll
  for (int i = 0; i < 2; ++i)
    #pragma unroll
    for (int j = 0; j < 2; ++j)
      #pragma unroll
      for (int r = 0; r < 8; ++r) acc[i][j].f[r] = 0.0f;

  // Stage one 32-wide K slice into buffer `buf` (3 b128 copies / thread).
  auto stage = [&](int buf, int kk) {
    #pragma unroll
    for (int q = 0; q < 2; ++q) {
      int row = srow + q * 64;
      copy_b128_to_lds(Xb + (size_t)(blockM + row) * II + kk + sseg * 8,
                       &As[buf][row * LDST + sseg * 8]);
    }
    copy_b128_to_lds(Ub + (size_t)(blockN + srow) * II + kk + sseg * 8,
                     &Bs[buf][srow * LDST + sseg * 8]);
  };

  const int NK = II / 32;              // 16 K-steps
  stage(0, 0);
  for (int ki = 0; ki < NK; ++ki) {
    const int buf = ki & 1;
    if (ki + 1 < NK) { stage(buf ^ 1, (ki + 1) * 32); wait_async<3>(); }
    else             { wait_async<0>(); }
    __syncthreads();                   // stage ki visible to all waves

    const unsigned short* Ab = As[buf];
    const unsigned short* Bb = Bs[buf];
    AB16 afrag[2], bfrag[2];
    #pragma unroll
    for (int i = 0; i < 2; ++i) {
      int r  = waveM * 32 + i * 16 + laneRow;
      int kb = laneHalf * 16;
      #pragma unroll
      for (int p = 0; p < 4; ++p) {
        afrag[i].u[p]     = *reinterpret_cast<const unsigned int*>(&Ab[r * LDST + kb + 2 * p]);
        afrag[i].u[p + 4] = *reinterpret_cast<const unsigned int*>(&Ab[r * LDST + 8 + kb + 2 * p]);
      }
    }
    #pragma unroll
    for (int j = 0; j < 2; ++j) {
      int c  = waveN * 32 + j * 16 + laneRow;
      int kb = laneHalf * 16;
      #pragma unroll
      for (int p = 0; p < 8; ++p)
        bfrag[j].u[p] = *reinterpret_cast<const unsigned int*>(&Bb[c * LDST + kb + 2 * p]);
    }
    #pragma unroll
    for (int i = 0; i < 2; ++i)
      #pragma unroll
      for (int j = 0; j < 2; ++j)
        acc[i][j].v = __builtin_amdgcn_wmma_f32_16x16x32_bf16(
            false, afrag[i].v, false, bfrag[j].v, (short)0, acc[i][j].v, false, false);
    __syncthreads();                   // done reading buf before it's restaged
  }

  #pragma unroll
  for (int i = 0; i < 2; ++i) {
    #pragma unroll
    for (int j = 0; j < 2; ++j) {
      int colg = blockN + waveN * 32 + j * 16 + laneRow;
      float bv = bias[colg];
      #pragma unroll
      for (int r = 0; r < 8; ++r) {
        int rowg = blockM + waveM * 32 + i * 16 + r + laneHalf * 8;
        out[(size_t)rowg * HH + colg] = acc[i][j].f[r] + bv;
      }
    }
  }
}

// ---------------------------------------------------------------------------
// Kernel 2: persistent recurrent scan. 32 co-resident blocks, each owns 32
// hidden columns, M = full batch 64. bf16 W tiles + bf16 h mirror staged via
// async-to-LDS (double-buffered); fp32 h kept for exact gate math.
// ---------------------------------------------------------------------------
__global__ __launch_bounds__(256)
void nbrc_scan_kernel(const float* __restrict__ xc,
                      const float* __restrict__ xa,
                      const float* __restrict__ xh,
                      const unsigned short* __restrict__ Wc16,
                      const unsigned short* __restrict__ Wa16,
                      float* __restrict__ hA, float* __restrict__ hB,
                      unsigned short* __restrict__ h16A,
                      unsigned short* __restrict__ h16B,
                      float* __restrict__ y,
                      float* __restrict__ hfinal,
                      unsigned int* __restrict__ bar)
{
  __shared__ unsigned short Hs[2][64 * LDST];
  __shared__ unsigned short WcS[2][32 * LDST];
  __shared__ unsigned short WaS[2][32 * LDST];

  const int tid   = threadIdx.x;
  const int lane  = tid & 31;
  const int wid   = tid >> 5;
  const int waveM = wid & 3;        // 4 M-tiles cover B=64
  const int waveN = wid >> 2;       // 2 N-tiles cover 32 cols
  const int laneHalf = lane >> 4;
  const int laneRow  = lane & 15;
  const int bn0 = blockIdx.x * 32;
  const unsigned int nBlocks = gridDim.x;
  const int srow = tid >> 2;        // 0..63
  const int sseg = tid & 3;

  for (int t = 0; t < TT; ++t) {
    const float* hsrc          = (t & 1) ? hB : hA;
    float* hdst                = (t & 1) ? hA : hB;
    const unsigned short* h16s = (t & 1) ? h16B : h16A;
    unsigned short* h16d       = (t & 1) ? h16A : h16B;

    C32 accC, accA;
    #pragma unroll
    for (int r = 0; r < 8; ++r) { accC.f[r] = 0.0f; accA.f[r] = 0.0f; }

    // 2 async b128 copies per thread per stage (waves 0-3: Wc, 4-7: Wa)
    auto stage = [&](int buf, int kk) {
      copy_b128_to_lds(h16s + (size_t)srow * HH + kk + sseg * 8,
                       &Hs[buf][srow * LDST + sseg * 8]);
      if (tid < 128) {
        int row = tid >> 2;
        copy_b128_to_lds(Wc16 + (size_t)(bn0 + row) * HH + kk + sseg * 8,
                         &WcS[buf][row * LDST + sseg * 8]);
      } else {
        int row = (tid - 128) >> 2;
        copy_b128_to_lds(Wa16 + (size_t)(bn0 + row) * HH + kk + sseg * 8,
                         &WaS[buf][row * LDST + sseg * 8]);
      }
    };

    const int NK = HH / 32;           // 32 K-steps
    stage(0, 0);
    for (int ki = 0; ki < NK; ++ki) {
      const int buf = ki & 1;
      if (ki + 1 < NK) { stage(buf ^ 1, (ki + 1) * 32); wait_async<2>(); }
      else             { wait_async<0>(); }
      __syncthreads();

      AB16 af, bc_, ba_;
      {
        int r  = waveM * 16 + laneRow;
        int kb = laneHalf * 16;
        #pragma unroll
        for (int p = 0; p < 4; ++p) {
          af.u[p]     = *reinterpret_cast<const unsigned int*>(&Hs[buf][r * LDST + kb + 2 * p]);
          af.u[p + 4] = *reinterpret_cast<const unsigned int*>(&Hs[buf][r * LDST + 8 + kb + 2 * p]);
        }
      }
      {
        int c  = waveN * 16 + laneRow;
        int kb = laneHalf * 16;
        #pragma unroll
        for (int p = 0; p < 8; ++p) {
          bc_.u[p] = *reinterpret_cast<const unsigned int*>(&WcS[buf][c * LDST + kb + 2 * p]);
          ba_.u[p] = *reinterpret_cast<const unsigned int*>(&WaS[buf][c * LDST + kb + 2 * p]);
        }
      }
      accC.v = __builtin_amdgcn_wmma_f32_16x16x32_bf16(false, af.v, false, bc_.v, (short)0, accC.v, false, false);
      accA.v = __builtin_amdgcn_wmma_f32_16x16x32_bf16(false, af.v, false, ba_.v, (short)0, accA.v, false, false);
      __syncthreads();
    }

    // Gates + state update (fp32), plus bf16 mirror for next step's matmul
    const size_t tbase = (size_t)t * (BB * HH);
    #pragma unroll
    for (int r = 0; r < 8; ++r) {
      int rowg = waveM * 16 + r + laneHalf * 8;   // batch index
      int colg = bn0 + waveN * 16 + laneRow;      // hidden index
      size_t idx = (size_t)rowg * HH + colg;
      float hp = hsrc[idx];
      float cg = 1.0f / (1.0f + __expf(-(xc[tbase + idx] + accC.f[r])));
      float ag = 1.0f + tanhf(xa[tbase + idx] + accA.f[r]);
      float hn = cg * hp + (1.0f - cg) * tanhf(xh[tbase + idx] + ag * hp);
      hdst[idx] = hn;
      h16d[idx] = f2bf(hn);
      y[tbase + idx] = hn;
      if (t == TT - 1) hfinal[idx] = hn;
    }

    // Device-scope grid barrier (monotonic counter, reset by memset each call)
    __threadfence();
    __syncthreads();
    if (tid == 0) {
      __hip_atomic_fetch_add(bar, 1u, __ATOMIC_RELEASE, __HIP_MEMORY_SCOPE_AGENT);
      unsigned int target = nBlocks * (unsigned int)(t + 1);
      while (__hip_atomic_load(bar, __ATOMIC_ACQUIRE, __HIP_MEMORY_SCOPE_AGENT) < target) {
        __builtin_amdgcn_s_sleep(1);
      }
    }
    __syncthreads();
  }
}

// ---------------------------------------------------------------------------
extern "C" void kernel_launch(void* const* d_in, const int* in_sizes, int n_in,
                              void* d_out, int out_size, void* d_ws, size_t ws_size,
                              hipStream_t stream) {
  (void)in_sizes; (void)n_in; (void)out_size; (void)ws_size;

  const float* x_seq = (const float*)d_in[0];
  const float* h0    = (const float*)d_in[1];
  const float* U_c   = (const float*)d_in[2];
  const float* W_c   = (const float*)d_in[3];
  const float* b_c   = (const float*)d_in[4];
  const float* U_a   = (const float*)d_in[5];
  const float* W_a   = (const float*)d_in[6];
  const float* b_a   = (const float*)d_in[7];
  const float* U_h   = (const float*)d_in[8];
  const float* b_h   = (const float*)d_in[9];

  float* y      = (float*)d_out;                  // [T,B,H]
  float* hfinal = y + (size_t)TT * BB * HH;       // [B,H]

  char* ws = (char*)d_ws;
  size_t off = 0;
  const size_t projBytes = (size_t)TT * BB * HH * sizeof(float);
  float* xc = (float*)(ws + off); off += projBytes;
  float* xa = (float*)(ws + off); off += projBytes;
  float* xh = (float*)(ws + off); off += projBytes;
  float* hA = (float*)(ws + off); off += (size_t)BB * HH * 4;
  float* hB = (float*)(ws + off); off += (size_t)BB * HH * 4;
  unsigned short* h16A = (unsigned short*)(ws + off); off += (size_t)BB * HH * 2;
  unsigned short* h16B = (unsigned short*)(ws + off); off += (size_t)BB * HH * 2;
  unsigned short* x16  = (unsigned short*)(ws + off); off += (size_t)TBM * II * 2;
  unsigned short* Uc16 = (unsigned short*)(ws + off); off += (size_t)HH * II * 2;
  unsigned short* Ua16 = (unsigned short*)(ws + off); off += (size_t)HH * II * 2;
  unsigned short* Uh16 = (unsigned short*)(ws + off); off += (size_t)HH * II * 2;
  unsigned short* Wc16 = (unsigned short*)(ws + off); off += (size_t)HH * HH * 2;
  unsigned short* Wa16 = (unsigned short*)(ws + off); off += (size_t)HH * HH * 2;
  unsigned int* bar = (unsigned int*)(ws + off);

  (void)hipMemsetAsync(bar, 0, sizeof(unsigned int), stream);
  (void)hipMemcpyAsync(hA, h0, (size_t)BB * HH * sizeof(float),
                       hipMemcpyDeviceToDevice, stream);

  // fp32 -> bf16 conversion passes
  auto cvt = [&](const float* src, unsigned short* dst, size_t nelem) {
    int npairs = (int)(nelem / 2);
    f32_to_bf16_kernel<<<dim3((npairs + 255) / 256), 256, 0, stream>>>(
        src, (unsigned int*)dst, npairs);
  };
  cvt(x_seq, x16,  (size_t)TBM * II);
  cvt(U_c,   Uc16, (size_t)HH * II);
  cvt(U_a,   Ua16, (size_t)HH * II);
  cvt(U_h,   Uh16, (size_t)HH * II);
  cvt(W_c,   Wc16, (size_t)HH * HH);
  cvt(W_a,   Wa16, (size_t)HH * HH);
  cvt(h0,    h16A, (size_t)BB * HH);

  dim3 pgrid(TBM / 128, HH / 64);
  proj_gemm_kernel<<<pgrid, 256, 0, stream>>>(x16, Uc16, b_c, xc);
  proj_gemm_kernel<<<pgrid, 256, 0, stream>>>(x16, Ua16, b_a, xa);
  proj_gemm_kernel<<<pgrid, 256, 0, stream>>>(x16, Uh16, b_h, xh);

  nbrc_scan_kernel<<<dim3(HH / 32), 256, 0, stream>>>(
      xc, xa, xh, Wc16, Wa16, hA, hB, h16A, h16B, y, hfinal, bar);
}
